// Bottleneck_32890859553321
// MI455X (gfx1250) — compile-verified, matrix operand order checked by reading, not run on hardware
//
#include <hip/hip_runtime.h>
#include <hip/hip_bf16.h>
#include <math.h>

typedef __attribute__((ext_vector_type(16))) _Float16 v16h;
typedef __attribute__((ext_vector_type(8)))  _Float16 v8h;
typedef __attribute__((ext_vector_type(4)))  _Float16 v4h;
typedef __attribute__((ext_vector_type(8)))  float    v8f;

#define NB 64     // batch
#define NP 128    // points
#define CD 128    // channels
#define FD 64     // interaction feature dim
#define NS 16     // neighbors
#define GD 16     // grouped channels (C / SHARE)

// WMMA 16x16x32 f16 per-lane K mapping: element e of lane (half = lane>=16)
//   k = (e<8 ? 0 : 16) + 8*half + (e&7)  -> two contiguous 8-element runs per fragment
__device__ __forceinline__ int kmap(int e, int half) {
  return ((e < 8) ? 0 : 16) + 8 * half + (e & 7);
}

__device__ __forceinline__ v16h frag_from_row(const _Float16* row, int ks, int half) {
  const v8h lo = *(const v8h*)(row + 32 * ks + 8 * half);
  const v8h hi = *(const v8h*)(row + 32 * ks + 16 + 8 * half);
  return __builtin_shufflevector(lo, hi, 0, 1, 2, 3, 4, 5, 6, 7,
                                 8, 9, 10, 11, 12, 13, 14, 15);
}

__device__ __forceinline__ float wsum(float x) {
#pragma unroll
  for (int o = 16; o > 0; o >>= 1) x += __shfl_xor(x, o, 32);
  return x;
}

__device__ __forceinline__ float gelu_exact(float x) {
  return 0.5f * x * (1.0f + erff(x * 0.70710678118654752f));
}

// ---------------- LN1: xn = LN(x); also f16 copy for WMMA ----------------
__global__ __launch_bounds__(32) void ln1_kernel(const float* __restrict__ x,
    const float* __restrict__ g, const float* __restrict__ b,
    float* __restrict__ xn, _Float16* __restrict__ xh)
{
  const size_t row = blockIdx.x;
  const int lane = threadIdx.x;
  const float4 xv = ((const float4*)(x + row * CD))[lane];
  const float mu = wsum(xv.x + xv.y + xv.z + xv.w) * (1.0f / CD);
  const float d0 = xv.x - mu, d1 = xv.y - mu, d2 = xv.z - mu, d3 = xv.w - mu;
  const float var = wsum(d0 * d0 + d1 * d1 + d2 * d2 + d3 * d3) * (1.0f / CD);
  const float rs = rsqrtf(var + 1e-5f);
  const int c = lane * 4;
  float4 o;
  o.x = d0 * rs * g[c + 0] + b[c + 0];
  o.y = d1 * rs * g[c + 1] + b[c + 1];
  o.z = d2 * rs * g[c + 2] + b[c + 2];
  o.w = d3 * rs * g[c + 3] + b[c + 3];
  ((float4*)(xn + row * CD))[lane] = o;
  v4h h;
  h[0] = (_Float16)o.x; h[1] = (_Float16)o.y;
  h[2] = (_Float16)o.z; h[3] = (_Float16)o.w;
  *(v4h*)(xh + row * CD + c) = h;
}

// ---- weight shuffle: f32 (KxN) -> f16 WMMA-fragment order, one v16h per lane ----
// dst layout: fragment (ks*(N/16)+ct) holds 32 lanes x 16 halves contiguous
__global__ __launch_bounds__(32) void shufw_kernel(const float* __restrict__ W,
                                                   _Float16* __restrict__ dst, int N)
{
  const int nt = N >> 4;
  const int ks = blockIdx.x / nt;
  const int ct = blockIdx.x - ks * nt;
  const int lane = threadIdx.x, half = lane >> 4, col = lane & 15;
  v16h b;
#pragma unroll
  for (int e = 0; e < 16; ++e)
    b[e] = (_Float16)W[(32 * ks + kmap(e, half)) * N + 16 * ct + col];
  ((v16h*)dst)[(size_t)blockIdx.x * 32 + lane] = b;
}

// ---------------- KNN: per point, top-16 smallest squared distance ----------------
__global__ void knn_kernel(const float* __restrict__ p, int* __restrict__ idx) {
  const int t = blockIdx.x * blockDim.x + threadIdx.x;
  if (t >= NB * NP) return;
  const int n = t >> 7;
  const int i = t & (NP - 1);
  const float4* pb = (const float4*)(p + (size_t)n * NP * 4);
  const float4 pi = pb[i];
  float bd[NS];
  int   bi[NS];
#pragma unroll
  for (int s = 0; s < NS; ++s) { bd[s] = 3.4e38f; bi[s] = 0; }
  for (int j = 0; j < NP; ++j) {
    const float4 pj = pb[j];
    const float e0 = pi.x - pj.x, e1 = pi.y - pj.y;
    const float e2 = pi.z - pj.z, e3 = pi.w - pj.w;
    const float dd = e0 * e0 + e1 * e1 + e2 * e2 + e3 * e3;
    if (dd < bd[NS - 1]) {
#pragma unroll
      for (int tp = NS - 1; tp >= 0; --tp) {
        const bool shift = (tp > 0) && (bd[tp - 1] > dd);
        if (shift)             { bd[tp] = bd[tp - 1]; bi[tp] = bi[tp - 1]; }
        else if (bd[tp] > dd)  { bd[tp] = dd;         bi[tp] = j; }
      }
    }
  }
#pragma unroll
  for (int s = 0; s < NS; ++s) idx[(size_t)t * NS + s] = bi[s];
}

// ---------------- q/k/v = xh @ W (shuffled weights, WMMA) ----------------
__global__ __launch_bounds__(32) void gemm_qkv_kernel(
    const _Float16* __restrict__ xh,
    const _Float16* __restrict__ wq, const _Float16* __restrict__ wk, const _Float16* __restrict__ wv,
    float* __restrict__ q, float* __restrict__ k, float* __restrict__ v)
{
  const int rt = blockIdx.x, n = blockIdx.y, which = blockIdx.z;
  const v16h* W = (const v16h*)((which == 0) ? wq : ((which == 1) ? wk : wv));
  float* out = (which == 0) ? q : ((which == 1) ? k : v);
  const int lane = threadIdx.x, half = lane >> 4, col = lane & 15;
  const _Float16* arow = xh + ((size_t)n * NP + 16 * rt + col) * CD;
  v16h a[4];
#pragma unroll
  for (int ks = 0; ks < 4; ++ks) a[ks] = frag_from_row(arow, ks, half);
  for (int ct = 0; ct < 8; ++ct) {
    v8f acc = {};
#pragma unroll
    for (int ks = 0; ks < 4; ++ks) {
      const v16h b = W[(ks * 8 + ct) * 32 + lane];
      acc = __builtin_amdgcn_wmma_f32_16x16x32_f16(false, a[ks], false, b, (short)0, acc, false, false);
    }
#pragma unroll
    for (int r = 0; r < 8; ++r)
      out[((size_t)n * NP + 16 * rt + r + 8 * half) * CD + 16 * ct + col] = acc[r];
  }
}

// ---------------- fused neighbor attention: one wave per (n, i) ----------------
__global__ __launch_bounds__(32) void attn_kernel(
    const float* __restrict__ q, const float* __restrict__ k, const float* __restrict__ v,
    const float* __restrict__ p, const float* __restrict__ U,
    const int* __restrict__ idx,
    const _Float16* __restrict__ wu, const _Float16* __restrict__ ww,
    const float* __restrict__ Wp, const float* __restrict__ bp,
    const float* __restrict__ bu, const float* __restrict__ bw,
    float* __restrict__ y)
{
  __shared__ int      sidx[NS];
  __shared__ float    sdp[NS][4];     // relative momentum (p_nb - p_i)
  __shared__ float    sk[NS][CD];     // gathered k rows
  __shared__ float    svv[NS][CD];    // v_nb, later v_nb + pe
  __shared__ _Float16 st[NS][CD];     // relu(q - k + pe + ue), f16 for logits WMMA
  __shared__ float    sattn[NS][GD];  // softmax weights

  const int i = blockIdx.x, n = blockIdx.y;
  const int lane = threadIdx.x, half = lane >> 4, col = lane & 15;

  if (lane < NS) {
    const int j = idx[((size_t)n * NP + i) * NS + lane];
    sidx[lane] = j;
    const float4 pj = ((const float4*)p)[(size_t)n * NP + j];
    const float4 pi = ((const float4*)p)[(size_t)n * NP + i];
    sdp[lane][0] = pj.x - pi.x;
    sdp[lane][1] = pj.y - pi.y;
    sdp[lane][2] = pj.z - pi.z;
    sdp[lane][3] = pj.w - pi.w;
  }
  __syncthreads();

  // cooperative stage of gathered k/v rows: one float4 per lane covers a 512B row
  for (int j = 0; j < NS; ++j) {
    const size_t roff = ((size_t)n * NP + sidx[j]) * CD;
    const float4 kv4 = ((const float4*)(k + roff))[lane];
    const float4 vv4 = ((const float4*)(v + roff))[lane];
    *(float4*)&sk[j][lane * 4]  = kv4;
    *(float4*)&svv[j][lane * 4] = vv4;
  }

  // A fragments: gathered U row (M = neighbor = lane&15), contiguous float4 loads + cvt
  v16h ua[2];
  {
    const float* urow = U + (((size_t)n * NP + i) * NP + sidx[col]) * FD;
#pragma unroll
    for (int ks = 0; ks < 2; ++ks) {
      const float* s0 = urow + 32 * ks + 8 * half;
      const float4 a0 = *(const float4*)(s0);
      const float4 a1 = *(const float4*)(s0 + 4);
      const float4 a2 = *(const float4*)(s0 + 16);
      const float4 a3 = *(const float4*)(s0 + 20);
      v16h t;
      t[0]  = (_Float16)a0.x; t[1]  = (_Float16)a0.y; t[2]  = (_Float16)a0.z; t[3]  = (_Float16)a0.w;
      t[4]  = (_Float16)a1.x; t[5]  = (_Float16)a1.y; t[6]  = (_Float16)a1.z; t[7]  = (_Float16)a1.w;
      t[8]  = (_Float16)a2.x; t[9]  = (_Float16)a2.y; t[10] = (_Float16)a2.z; t[11] = (_Float16)a2.w;
      t[12] = (_Float16)a3.x; t[13] = (_Float16)a3.y; t[14] = (_Float16)a3.z; t[15] = (_Float16)a3.w;
      ua[ks] = t;
    }
  }
  __syncthreads();

  const size_t qrow = ((size_t)n * NP + i) * CD;
  const v16h* wuv = (const v16h*)wu;   // fragment (ks*8+ct)

  // 8 column tiles of C: ue WMMA + elementwise fuse; stage t (f16), svv += pe (in place)
  for (int ct = 0; ct < 8; ++ct) {
    const int c = 16 * ct + col;
    v8f acc = {};
#pragma unroll
    for (int ks = 0; ks < 2; ++ks) {
      const v16h b = wuv[(ks * 8 + ct) * 32 + lane];
      acc = __builtin_amdgcn_wmma_f32_16x16x32_f16(false, ua[ks], false, b, (short)0, acc, false, false);
    }
    const float qv  = q[qrow + c];
    const float bpc = bp[c];
    const float buc = bu[c];
    const float w0 = Wp[0 * CD + c], w1 = Wp[1 * CD + c];
    const float w2 = Wp[2 * CD + c], w3 = Wp[3 * CD + c];
#pragma unroll
    for (int r = 0; r < 8; ++r) {
      const int j = r + 8 * half;
      const float pe = sdp[j][0] * w0 + sdp[j][1] * w1 + sdp[j][2] * w2 + sdp[j][3] * w3 + bpc;
      const float t = qv - sk[j][c] + pe + (acc[r] + buc);
      st[j][c]  = (_Float16)fmaxf(t, 0.0f);
      svv[j][c] += pe;
    }
  }
  __syncthreads();

  // logits = t(16xC) @ Ww(CxG): A from LDS (contiguous v8h runs), B from shuffled ww
  v8f lg = {};
  const v16h* wwv = (const v16h*)ww;   // fragment ks (nt = 1)
#pragma unroll
  for (int ks = 0; ks < 4; ++ks) {
    const v16h a = frag_from_row(&st[col][0], ks, half);
    const v16h b = wwv[ks * 32 + lane];
    lg = __builtin_amdgcn_wmma_f32_16x16x32_f16(false, a, false, b, (short)0, lg, false, false);
  }

  // softmax over neighbor axis (rows of 16x16 tile): 8 in-lane + partner lane (lane^16)
  const float bwg = bw[col];
  float lv[8];
  float mx = -3.4e38f;
#pragma unroll
  for (int r = 0; r < 8; ++r) { lv[r] = lg[r] + bwg; mx = fmaxf(mx, lv[r]); }
  mx = fmaxf(mx, __shfl_xor(mx, 16, 32));
  float s = 0.0f;
#pragma unroll
  for (int r = 0; r < 8; ++r) { lv[r] = __expf(lv[r] - mx); s += lv[r]; }
  s += __shfl_xor(s, 16, 32);
  const float inv = 1.0f / s;
#pragma unroll
  for (int r = 0; r < 8; ++r) sattn[r + 8 * half][col] = lv[r] * inv;
  __syncthreads();

  // y[c] = sum_j (v_nb+pe)[j][c] * attn[j][c % G]
#pragma unroll
  for (int cc = 0; cc < 4; ++cc) {
    const int c = lane * 4 + cc;
    const int g = c & (GD - 1);
    float accy = 0.0f;
#pragma unroll
    for (int j = 0; j < NS; ++j) accy += svv[j][c] * sattn[j][g];
    y[qrow + c] = accy;
  }
}

// ---------------- LN2 + exact gelu -> f16 for W3 GEMM ----------------
__global__ __launch_bounds__(32) void ln2gelu_kernel(const float* __restrict__ y,
    const float* __restrict__ g, const float* __restrict__ b, _Float16* __restrict__ yh)
{
  const size_t row = blockIdx.x;
  const int lane = threadIdx.x;
  const float4 xv = ((const float4*)(y + row * CD))[lane];
  const float mu = wsum(xv.x + xv.y + xv.z + xv.w) * (1.0f / CD);
  const float d0 = xv.x - mu, d1 = xv.y - mu, d2 = xv.z - mu, d3 = xv.w - mu;
  const float var = wsum(d0 * d0 + d1 * d1 + d2 * d2 + d3 * d3) * (1.0f / CD);
  const float rs = rsqrtf(var + 1e-5f);
  const int c = lane * 4;
  v4h h;
  h[0] = (_Float16)gelu_exact(d0 * rs * g[c + 0] + b[c + 0]);
  h[1] = (_Float16)gelu_exact(d1 * rs * g[c + 1] + b[c + 1]);
  h[2] = (_Float16)gelu_exact(d2 * rs * g[c + 2] + b[c + 2]);
  h[3] = (_Float16)gelu_exact(d3 * rs * g[c + 3] + b[c + 3]);
  *(v4h*)(yh + row * CD + c) = h;
}

// ---------------- y3 = yh @ W3 (shuffled weights, WMMA) ----------------
__global__ __launch_bounds__(32) void gemm3_kernel(
    const _Float16* __restrict__ yh, const _Float16* __restrict__ w3, float* __restrict__ y3)
{
  const int rt = blockIdx.x, n = blockIdx.y;
  const int lane = threadIdx.x, half = lane >> 4, col = lane & 15;
  const _Float16* arow = yh + ((size_t)n * NP + 16 * rt + col) * CD;
  const v16h* W = (const v16h*)w3;
  v16h a[4];
#pragma unroll
  for (int ks = 0; ks < 4; ++ks) a[ks] = frag_from_row(arow, ks, half);
  for (int ct = 0; ct < 8; ++ct) {
    v8f acc = {};
#pragma unroll
    for (int ks = 0; ks < 4; ++ks) {
      const v16h b = W[(ks * 8 + ct) * 32 + lane];
      acc = __builtin_amdgcn_wmma_f32_16x16x32_f16(false, a[ks], false, b, (short)0, acc, false, false);
    }
#pragma unroll
    for (int r = 0; r < 8; ++r)
      y3[((size_t)n * NP + 16 * rt + r + 8 * half) * CD + 16 * ct + col] = acc[r];
  }
}

// ---------------- LN3 + residual + gelu -> out ----------------
__global__ __launch_bounds__(32) void final_kernel(const float* __restrict__ y3,
    const float* __restrict__ g, const float* __restrict__ b,
    const float* __restrict__ xn, float* __restrict__ out)
{
  const size_t row = blockIdx.x;
  const int lane = threadIdx.x;
  const float4 xv = ((const float4*)(y3 + row * CD))[lane];
  const float mu = wsum(xv.x + xv.y + xv.z + xv.w) * (1.0f / CD);
  const float d0 = xv.x - mu, d1 = xv.y - mu, d2 = xv.z - mu, d3 = xv.w - mu;
  const float var = wsum(d0 * d0 + d1 * d1 + d2 * d2 + d3 * d3) * (1.0f / CD);
  const float rs = rsqrtf(var + 1e-5f);
  const float4 idv = ((const float4*)(xn + row * CD))[lane];
  const int c = lane * 4;
  float4 o;
  o.x = gelu_exact(d0 * rs * g[c + 0] + b[c + 0] + idv.x);
  o.y = gelu_exact(d1 * rs * g[c + 1] + b[c + 1] + idv.y);
  o.z = gelu_exact(d2 * rs * g[c + 2] + b[c + 2] + idv.z);
  o.w = gelu_exact(d3 * rs * g[c + 3] + b[c + 3] + idv.w);
  ((float4*)(out + row * CD))[lane] = o;
}

// ---------------- new_p = p (second tuple output) ----------------
__global__ void copyp_kernel(const float* __restrict__ p, float* __restrict__ out) {
  const int t = blockIdx.x * blockDim.x + threadIdx.x;
  if (t < NB * NP * 4) out[t] = p[t];
}

extern "C" void kernel_launch(void* const* d_in, const int* in_sizes, int n_in,
                              void* d_out, int out_size, void* d_ws, size_t ws_size,
                              hipStream_t stream) {
  (void)in_sizes; (void)n_in; (void)out_size; (void)ws_size;
  const float* x    = (const float*)d_in[0];
  const float* p    = (const float*)d_in[1];
  const float* U    = (const float*)d_in[2];
  const float* ln1g = (const float*)d_in[3];
  const float* ln1b = (const float*)d_in[4];
  const float* Wq   = (const float*)d_in[5];
  const float* Wk   = (const float*)d_in[6];
  const float* Wv   = (const float*)d_in[7];
  const float* Wp   = (const float*)d_in[8];
  const float* bp   = (const float*)d_in[9];
  const float* Wu   = (const float*)d_in[10];
  const float* bu   = (const float*)d_in[11];
  const float* Ww   = (const float*)d_in[12];
  const float* bw   = (const float*)d_in[13];
  const float* ln2g = (const float*)d_in[14];
  const float* ln2b = (const float*)d_in[15];
  const float* W3   = (const float*)d_in[16];
  const float* ln3g = (const float*)d_in[17];
  const float* ln3b = (const float*)d_in[18];
  float* out = (float*)d_out;

  // workspace layout (bytes): 6 f32 (N,P,C) buffers, 2 f16 copies, idx, shuffled f16 weights
  char* ws = (char*)d_ws;
  constexpr size_t ROWS = (size_t)NB * NP;   // 8192
  constexpr size_t XC   = ROWS * CD;         // 1,048,576 elems
  float*    xn  = (float*)(ws + 0 * XC);
  float*    qb  = (float*)(ws + 4 * XC);
  float*    kb  = (float*)(ws + 8 * XC);
  float*    vb  = (float*)(ws + 12 * XC);
  float*    yb  = (float*)(ws + 16 * XC);
  float*    y3  = (float*)(ws + 20 * XC);
  _Float16* xh  = (_Float16*)(ws + 24 * XC);
  _Float16* yh  = (_Float16*)(ws + 26 * XC);
  int*      idx = (int*)(ws + 28 * XC);
  _Float16* wqh = (_Float16*)(ws + 28 * XC + ROWS * NS * sizeof(int));
  _Float16* wkh = wqh + CD * CD;
  _Float16* wvh = wkh + CD * CD;
  _Float16* w3h = wvh + CD * CD;
  _Float16* wuh = w3h + CD * CD;
  _Float16* wwh = wuh + FD * CD;

  ln1_kernel<<<dim3((unsigned)ROWS), dim3(32), 0, stream>>>(x, ln1g, ln1b, xn, xh);
  // shuffle weights to WMMA fragment order: grid = (K/32) * (N/16)
  shufw_kernel<<<dim3(32), dim3(32), 0, stream>>>(Wq, wqh, CD);   // 128x128
  shufw_kernel<<<dim3(32), dim3(32), 0, stream>>>(Wk, wkh, CD);
  shufw_kernel<<<dim3(32), dim3(32), 0, stream>>>(Wv, wvh, CD);
  shufw_kernel<<<dim3(32), dim3(32), 0, stream>>>(W3, w3h, CD);
  shufw_kernel<<<dim3(16), dim3(32), 0, stream>>>(Wu, wuh, CD);   // 64x128
  shufw_kernel<<<dim3(4),  dim3(32), 0, stream>>>(Ww, wwh, GD);   // 128x16
  knn_kernel<<<dim3(32), dim3(256), 0, stream>>>(p, idx);
  gemm_qkv_kernel<<<dim3(8, NB, 3), dim3(32), 0, stream>>>(xh, wqh, wkh, wvh, qb, kb, vb);
  attn_kernel<<<dim3(NP, NB), dim3(32), 0, stream>>>(qb, kb, vb, p, U, idx, wuh, wwh, Wp, bp, bu, bw, yb);
  ln2gelu_kernel<<<dim3((unsigned)ROWS), dim3(32), 0, stream>>>(yb, ln2g, ln2b, yh);
  gemm3_kernel<<<dim3(8, NB), dim3(32), 0, stream>>>(yh, w3h, y3);
  final_kernel<<<dim3((unsigned)ROWS), dim3(32), 0, stream>>>(y3, ln3g, ln3b, xn, out);
  copyp_kernel<<<dim3(128), dim3(256), 0, stream>>>(p, out + XC);
}
